// ContrastiveLoss_8624294331097
// MI455X (gfx1250) — matrix operand order repaired and checked by reading, not compiled, not run on hardware
//
#include <hip/hip_runtime.h>

// ---------------------------------------------------------------------------
// Fused contrastive loss for MI455X (gfx1250, wave32, WMMA).
//   S = (E E^T)/0.1,  T = f f^T (f row-normalized)
//   bce = pos ? softplus(-S) : softplus(S) on off-diagonal, per-row mean,
//   averaged over rows having >=1 pos and >=1 neg.
//
// fused_loss: one block owns 32 rows (two 16-row WMMA sub-stripes that share
// every B fragment -> 2x arithmetic intensity vs L2), sweeps all 4096 cols.
// ---------------------------------------------------------------------------

#define BSZ   4096
#define DEMB  512
#define DFEAT 128
#define INV_TEMP 10.0f
#define THRESH   0.5f

typedef __bf16 bf16_t;
typedef __attribute__((ext_vector_type(8)))  __bf16 v8bf;
typedef __attribute__((ext_vector_type(16))) __bf16 v16bf;
typedef __attribute__((ext_vector_type(8)))  float  v8f;

__device__ __forceinline__ unsigned short f2bf(float f) {
  unsigned u = __float_as_uint(f);
  u += 0x7FFFu + ((u >> 16) & 1u);   // round-to-nearest-even
  return (unsigned short)(u >> 16);
}

// A-operand fragment (16-bit A 16x32 layout): lane group g = lane>>4 holds
// K = [8g..8g+7] in elements 0..7 and K = [8g+16..8g+23] in elements 8..15.
__device__ __forceinline__ v16bf load_a_frag(const bf16_t* p) {
  v8bf lo = *(const v8bf*)(p);
  v8bf hi = *(const v8bf*)(p + 16);
  return __builtin_shufflevector(lo, hi, 0,1,2,3,4,5,6,7,8,9,10,11,12,13,14,15);
}

// ---------------------------------------------------------------------------
// Prep kernels: fp32 -> bf16 (embeddings), L2-normalize + bf16 (features).
// ---------------------------------------------------------------------------
__global__ void __launch_bounds__(256) prep_emb(const float* __restrict__ e,
                                                unsigned short* __restrict__ out) {
  const int total4 = BSZ * DEMB / 4;
  for (int i = blockIdx.x * blockDim.x + threadIdx.x; i < total4;
       i += gridDim.x * blockDim.x) {
    float4 v = *(const float4*)(e + (size_t)i * 4);
    out[i * 4 + 0] = f2bf(v.x);
    out[i * 4 + 1] = f2bf(v.y);
    out[i * 4 + 2] = f2bf(v.z);
    out[i * 4 + 3] = f2bf(v.w);
  }
}

__global__ void __launch_bounds__(256) prep_feat(const float* __restrict__ f,
                                                 unsigned short* __restrict__ out) {
  const int wv = threadIdx.x >> 5;     // wave in block (8 waves)
  const int ln = threadIdx.x & 31;     // lane (wave32)
  const int row = blockIdx.x * 8 + wv; // one wave per row
  const size_t o = (size_t)row * DFEAT + ln * 4;
  float4 v = *(const float4*)(f + o);
  float ss = v.x * v.x + v.y * v.y + v.z * v.z + v.w * v.w;
#pragma unroll
  for (int m = 16; m > 0; m >>= 1) ss += __shfl_xor(ss, m, 32);
  const float scale = 1.0f / fmaxf(sqrtf(ss), 1e-12f);
  out[o + 0] = f2bf(v.x * scale);
  out[o + 1] = f2bf(v.y * scale);
  out[o + 2] = f2bf(v.z * scale);
  out[o + 3] = f2bf(v.w * scale);
}

// ---------------------------------------------------------------------------
// Fused dual-GEMM + BCE row reduction.
// Block = 32 rows (2 WMMA sub-stripes), 8 waves sweep 16-col tiles.
// ---------------------------------------------------------------------------
__global__ void __launch_bounds__(256) fused_loss(const unsigned short* __restrict__ embB_,
                                                  const unsigned short* __restrict__ fB_,
                                                  float* __restrict__ rowLoss,
                                                  int* __restrict__ rowValid) {
  constexpr int MROWS  = 32;
  constexpr int EMB_LD = DEMB + 8;   // pad 16B to dodge LDS bank conflicts
  constexpr int F_LD   = DFEAT + 8;
  __shared__ bf16_t   sEmb[MROWS][EMB_LD];
  __shared__ bf16_t   sF[MROWS][F_LD];
  __shared__ float    sBce[MROWS];
  __shared__ unsigned sPos[MROWS];

  const bf16_t* embB = (const bf16_t*)embB_;
  const bf16_t* fB   = (const bf16_t*)fB_;
  const int tid = threadIdx.x;
  const int rowBase = blockIdx.x * MROWS;

  if (tid < MROWS) { sBce[tid] = 0.0f; sPos[tid] = 0u; }

  // Stage the 32-row A-operand stripes into LDS (16B chunks).
  for (int c = tid; c < MROWS * 64; c += 256) {
    const int r = c >> 6, cc = c & 63;
    *(uint4*)&sEmb[r][cc * 8] =
        *(const uint4*)(embB + ((size_t)(rowBase + r) << 9) + cc * 8);
  }
  for (int c = tid; c < MROWS * 16; c += 256) {
    const int r = c >> 4, cc = c & 15;
    *(uint4*)&sF[r][cc * 8] =
        *(const uint4*)(fB + ((size_t)(rowBase + r) << 7) + cc * 8);
  }
  __syncthreads();

  const int wv  = tid >> 5;
  const int ln  = tid & 31;
  const int nn  = ln & 15;        // A row index == B column index (N)
  const int grp = ln >> 4;
  const int kbA = grp * 8;        // A fragment chunk base
  const int kbB = grp * 8;        // B fragment chunk base

  float bceAcc[2][8];
  int   posAcc[2][8];
#pragma unroll
  for (int s = 0; s < 2; ++s)
#pragma unroll
    for (int g = 0; g < 8; ++g) { bceAcc[s][g] = 0.0f; posAcc[s][g] = 0; }

  for (int colBase = wv * 16; colBase < BSZ; colBase += 128) {
    const int j = colBase + nn;
    const bf16_t* pBe = embB + (size_t)j * DEMB + kbB;
    const bf16_t* pBf = fB   + (size_t)j * DFEAT + kbB;
    if (colBase + 128 < BSZ)
      __builtin_prefetch(embB + (size_t)(j + 128) * DEMB, 0, 1);

    v8f accS0 = {0.f,0.f,0.f,0.f,0.f,0.f,0.f,0.f};
    v8f accS1 = {0.f,0.f,0.f,0.f,0.f,0.f,0.f,0.f};
    v8f accT0 = {0.f,0.f,0.f,0.f,0.f,0.f,0.f,0.f};
    v8f accT1 = {0.f,0.f,0.f,0.f,0.f,0.f,0.f,0.f};

    // S = E E^T : software-pipelined B fetch (next k-chunk issued before use).
    v16bf bCur = load_a_frag(pBe);
#pragma unroll
    for (int k = 0; k < DEMB; k += 32) {
      v16bf bNext = (k + 32 < DEMB) ? load_a_frag(pBe + k + 32) : bCur;
      v16bf a0 = load_a_frag(&sEmb[nn][k + kbA]);        // ds_load
      v16bf a1 = load_a_frag(&sEmb[16 + nn][k + kbA]);   // ds_load
      accS0 = __builtin_amdgcn_wmma_f32_16x16x32_bf16(
          false, a0, false, bCur, (short)0, accS0, false, false);
      accS1 = __builtin_amdgcn_wmma_f32_16x16x32_bf16(
          false, a1, false, bCur, (short)0, accS1, false, false);
      bCur = bNext;
    }

    // T = f f^T
    v16bf cCur = load_a_frag(pBf);
#pragma unroll
    for (int k = 0; k < DFEAT; k += 32) {
      v16bf cNext = (k + 32 < DFEAT) ? load_a_frag(pBf + k + 32) : cCur;
      v16bf a0 = load_a_frag(&sF[nn][k + kbA]);
      v16bf a1 = load_a_frag(&sF[16 + nn][k + kbA]);
      accT0 = __builtin_amdgcn_wmma_f32_16x16x32_bf16(
          false, a0, false, cCur, (short)0, accT0, false, false);
      accT1 = __builtin_amdgcn_wmma_f32_16x16x32_bf16(
          false, a1, false, cCur, (short)0, accT1, false, false);
      cCur = cNext;
    }

    // C layout: lane L, VGPR g -> M = g + 8*(L>>4), N = L&15.
#pragma unroll
    for (int s = 0; s < 2; ++s) {
#pragma unroll
      for (int g = 0; g < 8; ++g) {
        const int i = rowBase + s * 16 + g + 8 * grp;
        const float T = (s == 0) ? accT0[g] : accT1[g];
        const float S = ((s == 0) ? accS0[g] : accS1[g]) * INV_TEMP;
        const bool diag = (i == j);
        const bool pos  = (T > THRESH) && !diag;
        const float x   = pos ? -S : S;
        const float sp  = fmaxf(x, 0.0f) + log1pf(__expf(-fabsf(x)));
        if (!diag) { bceAcc[s][g] += sp; posAcc[s][g] += pos ? 1 : 0; }
      }
    }
  }

#pragma unroll
  for (int s = 0; s < 2; ++s)
#pragma unroll
    for (int g = 0; g < 8; ++g) {
      atomicAdd(&sBce[s * 16 + g + 8 * grp], bceAcc[s][g]);
      atomicAdd(&sPos[s * 16 + g + 8 * grp], (unsigned)posAcc[s][g]);
    }
  __syncthreads();

  if (tid < MROWS) {
    const unsigned p = sPos[tid];
    const bool valid = (p >= 1u) && (p <= (unsigned)(BSZ - 2));  // >=1 pos, >=1 neg
    rowLoss[rowBase + tid]  = valid ? (sBce[tid] * (1.0f / (BSZ - 1))) : 0.0f;
    rowValid[rowBase + tid] = valid ? 1 : 0;
  }
}

// ---------------------------------------------------------------------------
// Final scalar reduction: loss = sum(valid row_loss) / max(#valid, 1).
// ---------------------------------------------------------------------------
__global__ void __launch_bounds__(256) finalize(const float* __restrict__ rowLoss,
                                                const int* __restrict__ rowValid,
                                                float* __restrict__ out) {
  __shared__ float sSum[256];
  __shared__ int   sCnt[256];
  const int tid = threadIdx.x;
  float s = 0.0f;
  int   c = 0;
  for (int i = tid; i < BSZ; i += 256) { s += rowLoss[i]; c += rowValid[i]; }
  sSum[tid] = s;
  sCnt[tid] = c;
  __syncthreads();
  for (int o = 128; o > 0; o >>= 1) {
    if (tid < o) { sSum[tid] += sSum[tid + o]; sCnt[tid] += sCnt[tid + o]; }
    __syncthreads();
  }
  if (tid == 0) {
    const int n = sCnt[0] > 0 ? sCnt[0] : 1;
    out[0] = sSum[0] / (float)n;
  }
}

// ---------------------------------------------------------------------------
extern "C" void kernel_launch(void* const* d_in, const int* in_sizes, int n_in,
                              void* d_out, int out_size, void* d_ws, size_t ws_size,
                              hipStream_t stream) {
  (void)in_sizes; (void)n_in; (void)out_size; (void)ws_size;
  const float* emb  = (const float*)d_in[0];   // [4096, 512] fp32
  const float* feat = (const float*)d_in[1];   // [4096, 128] fp32

  unsigned short* embB = (unsigned short*)d_ws;            // 4096*512 bf16
  unsigned short* fB   = embB + (size_t)BSZ * DEMB;        // 4096*128 bf16
  float* rowLoss = (float*)(fB + (size_t)BSZ * DFEAT);     // 4096 f32
  int*   rowValid = (int*)(rowLoss + BSZ);                 // 4096 i32

  prep_emb<<<2048, 256, 0, stream>>>(emb, embB);
  prep_feat<<<BSZ / 8, 256, 0, stream>>>(feat, fB);
  fused_loss<<<BSZ / 32, 256, 0, stream>>>(embB, fB, rowLoss, rowValid);
  finalize<<<1, 256, 0, stream>>>(rowLoss, rowValid, (float*)d_out);
}
